// KNNModel_1099511627901
// MI455X (gfx1250) — matrix operand order repaired and checked by reading, not compiled, not run on hardware
//
#include <hip/hip_runtime.h>
#include <hip/hip_bf16.h>

// ---------------------------------------------------------------------------
// KNN multi-label classifier for MI455X (gfx1250, wave32, WMMA).
//   sims = norm(Q[512,128]) @ norm(M[500000,128])^T   -> bf16 WMMA GEMM
//   top-50 per query -> weighted label vote over 151 labels -> (>=0.3) int32
// ---------------------------------------------------------------------------

typedef __attribute__((ext_vector_type(16))) __bf16 v16bf;
typedef __attribute__((ext_vector_type(8)))  __bf16 v8bf;
typedef __attribute__((ext_vector_type(8)))  float  v8f;

#define N_MEM      500000
#define N_PAD      500032          // 64 * 7813
#define DIMS       128
#define NQ         512
#define NLAB       151
#define K_TOP      50
#define CHUNKS     64
#define CHUNK_ROWS 7813
#define QTILES     32
#define THRESH     0.3f

static __device__ inline unsigned short f2bf_bits(float f) {
    union { float f; unsigned u; } c; c.f = f;
    unsigned u = c.u;
    unsigned r = u + 0x7FFFu + ((u >> 16) & 1u);   // round to nearest even
    return (unsigned short)(r >> 16);
}

// Load the 4 B fragments (K = 0..127 in 4 steps of 32) for one 16-row tile.
// B (32x16) in this code's contiguous k order, matching A: lane%16 = N column (memory row),
// half-wave selects K range, VGPR v holds K=2v,2v+1 -> 16 contiguous bf16.
static __device__ inline void load_bfrags(v16bf bf[4], const __bf16* __restrict__ base) {
    #pragma unroll
    for (int kb = 0; kb < 4; kb++)
        bf[kb] = *(const v16bf*)(base + kb * 32);
}

// ---------------------------------------------------------------------------
// Kernel 1: L2-normalize rows of a [nrows,128] f32 matrix into bf16.
// Rows in [nrows, nrows_pad) are zero-filled. One wave per row, 8 rows/block.
// ---------------------------------------------------------------------------
__global__ __launch_bounds__(256) void normalize_bf16_kernel(
    const float* __restrict__ src, __bf16* __restrict__ dst,
    int nrows, int nrows_pad)
{
    int row  = blockIdx.x * 8 + (threadIdx.x >> 5);
    int lane = threadIdx.x & 31;
    if (row >= nrows_pad) return;

    uint2* drow = (uint2*)(dst + (size_t)row * DIMS);   // 32 lanes * 8B = 256B
    if (row >= nrows) {                                  // zero padding rows
        uint2 z; z.x = 0u; z.y = 0u;
        drow[lane] = z;
        return;
    }
    const float4* s = (const float4*)(src + (size_t)row * DIMS);
    float4 v = s[lane];
    float ss = v.x * v.x + v.y * v.y + v.z * v.z + v.w * v.w;
    #pragma unroll
    for (int off = 16; off > 0; off >>= 1) ss += __shfl_xor(ss, off, 32);
    float scale = 1.0f / fmaxf(sqrtf(ss), 1e-12f);

    unsigned short b0 = f2bf_bits(v.x * scale);
    unsigned short b1 = f2bf_bits(v.y * scale);
    unsigned short b2 = f2bf_bits(v.z * scale);
    unsigned short b3 = f2bf_bits(v.w * scale);
    uint2 o;
    o.x = (unsigned)b0 | ((unsigned)b1 << 16);
    o.y = (unsigned)b2 | ((unsigned)b3 << 16);
    drow[lane] = o;
}

// ---------------------------------------------------------------------------
// Kernel 2: bf16 WMMA GEMM + per-chunk top-50.
// Grid: (QTILES, CHUNKS)  -- qtile on x so the 32 co-resident blocks of one
// chunk reuse the same 2MB memory slice out of L2 (192MB); total GEMM HBM
// traffic ~= one pass over the 128MB bf16 matrix.
// Block: 128 threads = 4 waves; each wave owns a 16-row sub-stream and
// computes 16(query) x 16(row) sim tiles with v_wmma_f32_16x16x32_bf16,
// double-buffering the next tile's B fragments under the WMMA chain.
// ---------------------------------------------------------------------------
__global__ __launch_bounds__(128) void knn_topk_kernel(
    const __bf16* __restrict__ qbf, const __bf16* __restrict__ mbf,
    float* __restrict__ partV, int* __restrict__ partI)
{
    __shared__ __align__(16) __bf16 sA[16 * 136];     // 16 q rows, padded stride
    __shared__ float stage[4][16 * 17];               // per-wave 16x16 sim tile
    __shared__ float topV[16][4 * K_TOP];             // [query][wave*50 + k]
    __shared__ int   topI[16][4 * K_TOP];

    const int qtile = blockIdx.x;
    const int chunk = blockIdx.y;
    const int tid   = threadIdx.x;
    const int wave  = tid >> 5;
    const int lane  = tid & 31;

    // --- load & pad the 16x128 bf16 query tile into LDS (dword copy) ---
    {
        const unsigned* qsrc = (const unsigned*)(qbf + (size_t)(qtile * 16) * DIMS);
        unsigned* sAd = (unsigned*)sA;
        for (int i = tid; i < 16 * 64; i += 128) {
            int r = i >> 6, c = i & 63;
            sAd[r * 68 + c] = qsrc[i];
        }
    }
    // --- init per-(query,wave) top-50 lists ---
    if (lane < 16) {
        #pragma unroll 4
        for (int k = 0; k < K_TOP; k++) {
            topV[lane][wave * K_TOP + k] = -2.0f;     // below min cosine
            topI[lane][wave * K_TOP + k] = 0;
        }
    }
    __syncthreads();

    // --- build the 4 A fragments (16x32) in contiguous k order, matching B ---
    // lanes 0-15: M=lane, K = {0..15}; lanes 16-31: K = {16..31}
    const int m     = lane & 15;
    const int khalf = (lane < 16) ? 0 : 16;
    v16bf afrag[4];
    #pragma unroll
    for (int kb = 0; kb < 4; kb++) {
        const v8bf lo = *(const v8bf*)&sA[m * 136 + kb * 32 + khalf];
        const v8bf hi = *(const v8bf*)&sA[m * 136 + kb * 32 + khalf + 8];
        afrag[kb] = __builtin_shufflevector(lo, hi,
            0, 1, 2, 3, 4, 5, 6, 7, 8, 9, 10, 11, 12, 13, 14, 15);
    }

    const int chunkStart = chunk * CHUNK_ROWS;
    const int chunkEnd   = chunkStart + CHUNK_ROWS;   // == N_PAD on last chunk
    const int colOff     = (lane & 15) * DIMS + ((lane < 16) ? 0 : 16);

    float mn = -2.0f;   // current min of my top-50 list (lane<16 only)
    int   mp = 0;
    float* myList = &topV[m][wave * K_TOP];
    int*   myIdx  = &topI[m][wave * K_TOP];
    float* st     = stage[wave];

    int rowbase = chunkStart + wave * 16;
    v16bf bfr[4];
    load_bfrags(bfr, mbf + (size_t)rowbase * DIMS + colOff);   // prologue

    for (; rowbase < chunkEnd; rowbase += 64) {
        const int nextbase = rowbase + 64;
        const bool hasNext = nextbase < chunkEnd;

        // Issue next tile's B loads first: they retire under the WMMA chain
        // and the top-k scan below (double buffer).
        v16bf bnx[4];
        if (hasNext) {
            const __bf16* nb = mbf + (size_t)nextbase * DIMS + colOff;
            __builtin_prefetch(nb + 64 * DIMS, 0, 3);   // 2 iterations ahead
            load_bfrags(bnx, nb);
        }

        v8f acc = {0.f, 0.f, 0.f, 0.f, 0.f, 0.f, 0.f, 0.f};
        #pragma unroll
        for (int kb = 0; kb < 4; kb++) {
            acc = __builtin_amdgcn_wmma_f32_16x16x32_bf16(
                false, afrag[kb], false, bfr[kb], (short)0, acc, false, false);
        }

        // C/D layout: VGPR j -> M = j (+8 for lanes 16-31), N = lane&15
        #pragma unroll
        for (int j = 0; j < 8; j++) {
            int mm = j + ((lane < 16) ? 0 : 8);
            st[mm * 17 + (lane & 15)] = acc[j];
        }
        // wave-internal LDS ops are in-order: safe to read the staged tile
        if (lane < 16) {
            #pragma unroll 4
            for (int col = 0; col < 16; col++) {
                float v = st[m * 17 + col];
                int  gr = rowbase + col;
                if (gr < N_MEM && v > mn) {
                    myList[mp] = v; myIdx[mp] = gr;
                    mn = myList[0]; mp = 0;
                    for (int k = 1; k < K_TOP; k++) {
                        float x = myList[k];
                        if (x < mn) { mn = x; mp = k; }
                    }
                }
            }
        }
        if (hasNext) {
            #pragma unroll
            for (int kb = 0; kb < 4; kb++) bfr[kb] = bnx[kb];
        }
    }
    __syncthreads();

    // --- merge 4x50 -> top-50 per query; each wave handles 4 queries ---
    for (int qq = 0; qq < 4; qq++) {
        int q  = wave * 4 + qq;
        int qg = qtile * 16 + q;
        float* vlist = topV[q];
        int*   ilist = topI[q];
        for (int k = 0; k < K_TOP; k++) {
            float bv = -1e30f; int bp = 0;
            for (int e = lane; e < 4 * K_TOP; e += 32) {
                float v = vlist[e];
                if (v > bv) { bv = v; bp = e; }
            }
            #pragma unroll
            for (int off = 16; off > 0; off >>= 1) {
                float ov = __shfl_xor(bv, off, 32);
                int   op = __shfl_xor(bp, off, 32);
                if (ov > bv) { bv = ov; bp = op; }
            }
            if (lane == 0) {
                size_t o = ((size_t)qg * CHUNKS + chunk) * K_TOP + k;
                partV[o] = bv;
                partI[o] = ilist[bp];
                vlist[bp] = -1e30f;   // remove; in-order LDS within wave
            }
        }
    }
}

// ---------------------------------------------------------------------------
// Kernel 3: per query, merge 64x50 candidates -> global top-50, then the
// weighted label vote and threshold. One block (256 thr) per query.
// ---------------------------------------------------------------------------
__global__ __launch_bounds__(256) void knn_final_kernel(
    const float* __restrict__ partV, const int* __restrict__ partI,
    const float* __restrict__ labels, int* __restrict__ out)
{
    __shared__ float cv[CHUNKS * K_TOP];    // 3200 candidates
    __shared__ int   ci[CHUNKS * K_TOP];
    __shared__ float selV[K_TOP];
    __shared__ int   selI[K_TOP];
    __shared__ float wsum_s;

    const int q   = blockIdx.x;
    const int tid = threadIdx.x;

    for (int i = tid; i < CHUNKS * K_TOP; i += 256) {
        size_t o = (size_t)q * CHUNKS * K_TOP + i;
        cv[i] = partV[o];
        ci[i] = partI[o];
    }
    __syncthreads();

    if (tid < 32) {            // wave 0: extract top-50 by repeated argmax
        const int lane = tid;
        float wsum = 0.0f;
        for (int k = 0; k < K_TOP; k++) {
            float bv = -1e30f; int bp = 0;
            for (int e = lane; e < CHUNKS * K_TOP; e += 32) {
                float v = cv[e];
                if (v > bv) { bv = v; bp = e; }
            }
            #pragma unroll
            for (int off = 16; off > 0; off >>= 1) {
                float ov = __shfl_xor(bv, off, 32);
                int   op = __shfl_xor(bp, off, 32);
                if (ov > bv) { bv = ov; bp = op; }
            }
            if (lane == 0) {
                selV[k] = bv;
                selI[k] = ci[bp];
                cv[bp] = -1e30f;
                wsum += bv;      // GAMMA == 1 -> weight = sim
            }
        }
        if (lane == 0) wsum_s = wsum + 1e-8f;
    }
    __syncthreads();

    const float wsum = wsum_s;
    if (tid < NLAB) {
        float acc = 0.0f;
        for (int k = 0; k < K_TOP; k++)
            acc += selV[k] * labels[(size_t)selI[k] * NLAB + tid];
        out[(size_t)q * NLAB + tid] = ((acc / wsum) >= THRESH) ? 1 : 0;
    }
}

// ---------------------------------------------------------------------------
// Launch
// ---------------------------------------------------------------------------
extern "C" void kernel_launch(void* const* d_in, const int* in_sizes, int n_in,
                              void* d_out, int out_size, void* d_ws, size_t ws_size,
                              hipStream_t stream) {
    const float* qf     = (const float*)d_in[0];   // [512,128]
    const float* mf     = (const float*)d_in[1];   // [500000,128]
    const float* labels = (const float*)d_in[2];   // [500000,151]
    int* out = (int*)d_out;                        // [512,151] int32

    char* ws = (char*)d_ws;
    __bf16* qbf = (__bf16*)ws;                                 // 131,072 B
    size_t off = (size_t)NQ * DIMS * 2;
    __bf16* mbf = (__bf16*)(ws + off);                         // 128,008,192 B
    off += (size_t)N_PAD * DIMS * 2;
    float* partV = (float*)(ws + off);                         // 6,553,600 B
    off += (size_t)NQ * CHUNKS * K_TOP * 4;
    int* partI = (int*)(ws + off);                             // 6,553,600 B

    normalize_bf16_kernel<<<NQ / 8, 256, 0, stream>>>(qf, qbf, NQ, NQ);
    normalize_bf16_kernel<<<N_PAD / 8, 256, 0, stream>>>(mf, mbf, N_MEM, N_PAD);
    knn_topk_kernel<<<dim3(QTILES, CHUNKS), 128, 0, stream>>>(qbf, mbf, partV, partI);
    knn_final_kernel<<<NQ, 256, 0, stream>>>(partV, partI, labels, out);
}
